// AttentionBlock_11690900979720
// MI455X (gfx1250) — compile-verified, matrix operand order checked by reading, not run on hardware
//
#include <hip/hip_runtime.h>

typedef __attribute__((ext_vector_type(16))) _Float16     v16h;
typedef __attribute__((ext_vector_type(8)))  float        v8f;
typedef __attribute__((ext_vector_type(4)))  unsigned int u32x4;
typedef __attribute__((ext_vector_type(8)))  int          i32x8;
typedef __attribute__((ext_vector_type(4)))  int          i32x4;

#define Cn   256
#define Nsp  4096      // H*W
#define Bn   4
#define GRP  16
#define FRAG 512       // halfs per packed fragment: 32 lanes x 16 halfs (32B/lane)

static __device__ inline v8f wmma_f16(v16h a, v16h b, v8f c) {
  return __builtin_amdgcn_wmma_f32_16x16x32_f16(
      false, a, false, b, (short)0, c, false, false);
}

// Packed-fragment addressing (matches ISA 16-bit A/B 16x32 striping):
// element (K in 0..31, col in 0..15) lives at lane = col + 16*((K>>3)&1),
// slot = (K&16 ? 8 : 0) + (K&7). Consumer lane l reads 16 halfs at l*16.
static __device__ inline int frag_slot(int Kin) {
  return ((Kin & 16) ? 8 : 0) + (Kin & 7);
}
static __device__ inline int frag_lane(int Kin, int col) {
  return col + (((Kin >> 3) & 1) << 4);
}
static __device__ inline v16h load_frag(const _Float16* p) {
  return *(const v16h*)p;            // 32B -> 2x b128 loads
}

// ---------------------------------------------------------------------------
// Tensor Data Mover: 1-D contiguous copy of `nelem` fp16 elements to LDS.
// Descriptor per CDNA5 ISA ch.8 (D# groups). clang-23 6-arg builtin:
//   (u32x4 g0, i32x8 g1, i32x4 g2, i32x4 g3, i32x8 extra, i32 cpol)
// ---------------------------------------------------------------------------
static __device__ inline unsigned lds_off(const void* p) {
  // flat LDS address = {aperture, offset32}; low 32 bits = LDS byte offset
  return (unsigned)(uintptr_t)p;
}
static __device__ inline void tdm_copy_to_lds(const _Float16* gsrc,
                                              unsigned lds_byte_off,
                                              unsigned nelem) {
  unsigned long long ga = (unsigned long long)(uintptr_t)gsrc;
  u32x4 g0;
  g0[0] = 1u;                                   // count=1 valid user descriptor
  g0[1] = lds_byte_off;                         // lds_addr (bytes)
  g0[2] = (unsigned)ga;                         // global_addr[31:0]
  g0[3] = (unsigned)((ga >> 32) & 0x1FFFFFFu)   // global_addr[56:32]
        | (2u << 30);                           // type = 2 ("image")
  i32x8 g1;
  g1[0] = (int)(1u << 16);                      // data_size=1 (2B); mask=0
  g1[1] = (int)((nelem & 0xFFFFu) << 16);       // tensor_dim0[15:0] @ bits63:48
  g1[2] = (int)((nelem >> 16) & 0xFFFFu)        // tensor_dim0[31:16] @ bits79:64
        | (int)(1u << 16);                      // tensor_dim1 = 1
  g1[3] = (int)((nelem & 0xFFFFu) << 16);       // tile_dim0 @ bits127:112
  g1[4] = 1;                                    // tile_dim1 = 1, tile_dim2 = 0
  g1[5] = (int)nelem;                           // tensor_dim0_stride lo32
  g1[6] = 0;
  g1[7] = 0;
  i32x4 z4 = {0, 0, 0, 0};
  i32x8 z8 = {0, 0, 0, 0, 0, 0, 0, 0};
  __builtin_amdgcn_tensor_load_to_lds(g0, g1, z4, z4, z8, 0);
}

// ---------------------------------------------------------------------------
// Kernel 0: weights fp32 -> packed fp16 fragments.
//   wq/wk/wv as A-frags [ot][kt]; Wo^T as B-frags [kt][ot].
// grid (8 kt, 16 ot), block 32.
// ---------------------------------------------------------------------------
__global__ __launch_bounds__(32) void prep_weights_kernel(
    const float* __restrict__ wq, const float* __restrict__ wk,
    const float* __restrict__ wv, const float* __restrict__ wo,
    _Float16* __restrict__ wqP, _Float16* __restrict__ wkP,
    _Float16* __restrict__ wvP, _Float16* __restrict__ woP) {
  const int kt = blockIdx.x, ot = blockIdx.y;
  const int lane = threadIdx.x;
  const int m = lane & 15, koff = (lane >> 4) * 8;
  const int row = ot * 16 + m;
  const size_t pa = (size_t)(ot * 8 + kt) * FRAG + lane * 16;   // A-frag slot
  const size_t pb = (size_t)(kt * 16 + ot) * FRAG + lane * 16;  // B-frag slot
#pragma unroll
  for (int i = 0; i < 8; ++i) {
    const int K0 = kt * 32 + koff + i;
    const int K1 = kt * 32 + 16 + koff + i;
    wqP[pa + i]     = (_Float16)wq[row * Cn + K0];
    wqP[pa + i + 8] = (_Float16)wq[row * Cn + K1];
    wkP[pa + i]     = (_Float16)wk[row * Cn + K0];
    wkP[pa + i + 8] = (_Float16)wk[row * Cn + K1];
    wvP[pa + i]     = (_Float16)wv[row * Cn + K0];
    wvP[pa + i + 8] = (_Float16)wv[row * Cn + K1];
    // B-frag of Wo^T: element (K=c, col=o) = wo[o*C + c]  (same gather shape)
    woP[pb + i]     = (_Float16)wo[row * Cn + K0];
    woP[pb + i + 8] = (_Float16)wo[row * Cn + K1];
  }
}

// ---------------------------------------------------------------------------
// Kernel 1: GroupNorm -> xn packed directly as B-frags [b][nt][kt].
// one block per (b, group). grid 64, block 256.
// ---------------------------------------------------------------------------
__global__ __launch_bounds__(256) void gn_kernel(const float* __restrict__ x,
                                                 const float* __restrict__ gamma,
                                                 const float* __restrict__ beta,
                                                 _Float16* __restrict__ xnP) {
  const int b = blockIdx.x >> 4, grp = blockIdx.x & 15, t = threadIdx.x;
  const int CPG = Cn / GRP;                  // 16 channels per group
  const int total = CPG * Nsp;               // 65536
  const size_t base = ((size_t)b * Cn + (size_t)grp * CPG) * Nsp;

  float s = 0.f, ss = 0.f;
  for (int e = t; e < total; e += 256) {
    float v = x[base + e];
    s += v; ss += v * v;
  }
  __shared__ float rs[256], rss[256];
  rs[t] = s; rss[t] = ss;
  __syncthreads();
  for (int off = 128; off > 0; off >>= 1) {
    if (t < off) { rs[t] += rs[t + off]; rss[t] += rss[t + off]; }
    __syncthreads();
  }
  const float inv_n = 1.0f / (float)total;
  const float mean  = rs[0] * inv_n;
  const float rstd  = rsqrtf(rss[0] * inv_n - mean * mean + 1e-5f);

  for (int e = t; e < total; e += 256) {
    const int c = grp * CPG + e / Nsp;       // global channel
    const int n = e & (Nsp - 1);             // spatial index
    const float v = (x[base + e] - mean) * rstd * gamma[c] + beta[c];
    const int nt = n >> 4, col = n & 15, kt = c >> 5, Kin = c & 31;
    xnP[((size_t)(b * 256 + nt) * 8 + kt) * FRAG +
        frag_lane(Kin, col) * 16 + frag_slot(Kin)] = (_Float16)v;
  }
}

// ---------------------------------------------------------------------------
// Kernel 2: QKV GEMMs, fully packed operands. One wave / 16x16 tile.
//   writes qP & kP (K=channels; identical packing) and vP (K=spatial).
// grid (N/16, C/16, B), block 32.
// ---------------------------------------------------------------------------
__global__ __launch_bounds__(32) void qkv_kernel(
    const _Float16* __restrict__ xnP,
    const _Float16* __restrict__ wqP, const _Float16* __restrict__ wkP,
    const _Float16* __restrict__ wvP,
    const float* __restrict__ bq, const float* __restrict__ bk,
    const float* __restrict__ bv,
    _Float16* __restrict__ qP, _Float16* __restrict__ kP,
    _Float16* __restrict__ vP) {
  const int b = blockIdx.z, ot = blockIdx.y, nt = blockIdx.x;
  const int lane = threadIdx.x, nl = lane & 15, hh = lane >> 4;

  v8f aq = {}, ak = {}, av = {};
#pragma unroll
  for (int kt = 0; kt < 8; ++kt) {
    const v16h fq = load_frag(wqP + (size_t)(ot * 8 + kt) * FRAG + lane * 16);
    const v16h fk = load_frag(wkP + (size_t)(ot * 8 + kt) * FRAG + lane * 16);
    const v16h fv = load_frag(wvP + (size_t)(ot * 8 + kt) * FRAG + lane * 16);
    const v16h xb = load_frag(xnP + ((size_t)(b * 256 + nt) * 8 + kt) * FRAG + lane * 16);
    aq = wmma_f16(fq, xb, aq);
    ak = wmma_f16(fk, xb, ak);
    av = wmma_f16(fv, xb, av);
  }

#pragma unroll
  for (int v = 0; v < 8; ++v) {
    const int m = v + hh * 8;                  // channel row within tile
    const int o = ot * 16 + m;
    const _Float16 qv = (_Float16)(aq[v] + bq[o]);
    const _Float16 kv = (_Float16)(ak[v] + bk[o]);
    const _Float16 vv = (_Float16)(av[v] + bv[o]);
    // q/k: K = channel o, col = spatial nl  (A- and B-frag pack identically)
    const int Kc = o & 31, ktc = o >> 5;
    const size_t fqk = ((size_t)(b * 256 + nt) * 8 + ktc) * FRAG +
                       frag_lane(Kc, nl) * 16 + frag_slot(Kc);
    qP[fqk] = qv;
    kP[fqk] = kv;
    // v: K = spatial j, col = channel m within tile ct = ot
    const int jg = nt * 16 + nl, Kj = jg & 31, jt32 = jg >> 5;
    vP[((size_t)(b * 128 + jt32) * 16 + ot) * FRAG +
       frag_lane(Kj, m) * 16 + frag_slot(Kj)] = vv;
  }
}

// ---------------------------------------------------------------------------
// Kernel 3: flash attention. 8 waves/block (4 query tiles x 2 channel halves)
// share K/V j-tiles staged in LDS by the Tensor Data Mover, double-buffered.
// grid (N/64, B), block 256. LDS: 2*16KB K + 2*16KB V + 8*1KB P = 72 KB.
// ---------------------------------------------------------------------------
__global__ __launch_bounds__(256) void attn_kernel(
    const _Float16* __restrict__ qP, const _Float16* __restrict__ kP,
    const _Float16* __restrict__ vP, _Float16* __restrict__ attP) {
  const int b = blockIdx.y;
  const int wave = threadIdx.x >> 5, lane = threadIdx.x & 31;
  const int it = blockIdx.x * 4 + (wave >> 1);   // query tile (16 rows)
  const int ch = wave & 1;                       // channel half (128 ch)
  const int nl = lane & 15, hh = lane >> 4, koff = hh * 8;

  __shared__ __align__(16) _Float16 sK[2][8192];  // 32 keys x 256 ch, 2 bufs
  __shared__ __align__(16) _Float16 sV[2][8192];  // 32 keys x 256 ch, 2 bufs
  __shared__ __align__(16) _Float16 sP[8][FRAG];  // per-wave prob staging

  // resident q fragments (all 256 channels of this query tile)
  v16h qa[8];
#pragma unroll
  for (int kt = 0; kt < 8; ++kt)
    qa[kt] = load_frag(qP + ((size_t)(b * 256 + it) * 8 + kt) * FRAG + lane * 16);

  v8f acc[8];
#pragma unroll
  for (int t = 0; t < 8; ++t) acc[t] = (v8f){};
  float rowM[8], rowL[8];
#pragma unroll
  for (int v = 0; v < 8; ++v) { rowM[v] = -3.0e38f; rowL[v] = 0.0f; }

  const float sc = 0.0625f * 1.44269504088896f;   // 1/sqrt(C) * log2(e)

  // prologue: TDM stage j-tile 0 into buffer 0
  if (threadIdx.x < 32) {
    tdm_copy_to_lds(kP + (size_t)(b * 256) * 8 * FRAG, lds_off(&sK[0][0]), 8192);
    tdm_copy_to_lds(vP + (size_t)(b * 128) * 16 * FRAG, lds_off(&sV[0][0]), 8192);
    __builtin_amdgcn_s_wait_tensorcnt(0);
  }
  __syncthreads();

  for (int j0 = 0; j0 < Nsp; j0 += 32) {
    const int buf = (j0 >> 5) & 1;
    if (threadIdx.x < 32 && (j0 + 32) < Nsp) {    // TDM prefetch next tile
      const int jn = j0 + 32;
      tdm_copy_to_lds(kP + (size_t)(b * 256 + (jn >> 4)) * 8 * FRAG,
                      lds_off(&sK[buf ^ 1][0]), 8192);
      tdm_copy_to_lds(vP + (size_t)(b * 128 + (jn >> 5)) * 16 * FRAG,
                      lds_off(&sV[buf ^ 1][0]), 8192);
    }

    // ---- scores: two 16x16 tiles over 32 keys (K frags from LDS) ----
    v8f s0 = {}, s1 = {};
#pragma unroll
    for (int kt = 0; kt < 8; ++kt) {
      const v16h k0 = *(const v16h*)(&sK[buf][(kt)     * FRAG + lane * 16]);
      const v16h k1 = *(const v16h*)(&sK[buf][(8 + kt) * FRAG + lane * 16]);
      s0 = wmma_f16(qa[kt], k0, s0);
      s1 = wmma_f16(qa[kt], k1, s1);
    }

    // ---- online softmax ----
    float alpha[8];
#pragma unroll
    for (int v = 0; v < 8; ++v) {
      const float a0 = s0[v] * sc, a1 = s1[v] * sc;
      float mx = fmaxf(a0, a1);
      mx = fmaxf(mx, __shfl_xor(mx, 1, 32));
      mx = fmaxf(mx, __shfl_xor(mx, 2, 32));
      mx = fmaxf(mx, __shfl_xor(mx, 4, 32));
      mx = fmaxf(mx, __shfl_xor(mx, 8, 32));
      const float newM = fmaxf(rowM[v], mx);
      const float al = exp2f(rowM[v] - newM);
      const float p0 = exp2f(a0 - newM);
      const float p1 = exp2f(a1 - newM);
      float rsum = p0 + p1;
      rsum += __shfl_xor(rsum, 1, 32);
      rsum += __shfl_xor(rsum, 2, 32);
      rsum += __shfl_xor(rsum, 4, 32);
      rsum += __shfl_xor(rsum, 8, 32);
      rowL[v] = rowL[v] * al + rsum;
      rowM[v] = newM;
      alpha[v] = al;
      const int row = v + hh * 8;
      sP[wave][row * 32 + nl]      = (_Float16)p0;
      sP[wave][row * 32 + 16 + nl] = (_Float16)p1;
    }
    __syncthreads();

    v16h pa;                       // probabilities in A-frag layout
#pragma unroll
    for (int i = 0; i < 8; ++i) {
      pa[i]     = sP[wave][nl * 32 + koff + i];
      pa[i + 8] = sP[wave][nl * 32 + 16 + koff + i];
    }
    __syncthreads();

    // ---- p @ V: 8 channel tiles for this wave's half (V frags from LDS) ----
#pragma unroll
    for (int t = 0; t < 8; ++t) {
      const v16h vf = *(const v16h*)(&sV[buf][(ch * 8 + t) * FRAG + lane * 16]);
#pragma unroll
      for (int v = 0; v < 8; ++v) acc[t][v] *= alpha[v];
      acc[t] = wmma_f16(pa, vf, acc[t]);
    }

    if (threadIdx.x < 32) __builtin_amdgcn_s_wait_tensorcnt(0);
    __syncthreads();               // hand off buffers
  }

  // ---- finalize & scatter-store attP as A-frags for the out-projection ----
#pragma unroll
  for (int t = 0; t < 8; ++t) {
#pragma unroll
    for (int v = 0; v < 8; ++v) {
      const int m = v + hh * 8;                    // query row within tile
      const int c = ch * 128 + t * 16 + nl;        // global channel (= K)
      const float o = acc[t][v] / rowL[v];
      attP[((size_t)(b * 256 + it) * 8 + (c >> 5)) * FRAG +
           frag_lane(c & 31, m) * 16 + frag_slot(c & 31)] = (_Float16)o;
    }
  }
}

// ---------------------------------------------------------------------------
// Kernel 4: output projection + residual (fp32 out), packed operands.
// grid (N/16, C/16, B), block 32.
// ---------------------------------------------------------------------------
__global__ __launch_bounds__(32) void proj_kernel(
    const _Float16* __restrict__ attP, const _Float16* __restrict__ woP,
    const float* __restrict__ bo, const float* __restrict__ x,
    float* __restrict__ out) {
  const int b = blockIdx.z, ot = blockIdx.y, nt = blockIdx.x;
  const int lane = threadIdx.x, nl = lane & 15, hh = lane >> 4;

  v8f acc = {};
#pragma unroll
  for (int kt = 0; kt < 8; ++kt) {
    const v16h fa = load_frag(attP + ((size_t)(b * 256 + nt) * 8 + kt) * FRAG + lane * 16);
    const v16h fb = load_frag(woP + (size_t)(kt * 16 + ot) * FRAG + lane * 16);
    acc = wmma_f16(fa, fb, acc);
  }
#pragma unroll
  for (int v = 0; v < 8; ++v) {
    const int m = v + hh * 8;                      // spatial row within tile
    const size_t idx = ((size_t)b * Cn + ot * 16 + nl) * Nsp + nt * 16 + m;
    out[idx] = acc[v] + bo[ot * 16 + nl] + x[idx];
  }
}

// ---------------------------------------------------------------------------
// Launcher. Workspace (all fp16, packed-fragment layouts), ~42.5 MB:
//   xnP | qP | kP | vP | attP  (B*C*N each)  +  wqP|wkP|wvP|woP (C*C each)
// ---------------------------------------------------------------------------
extern "C" void kernel_launch(void* const* d_in, const int* in_sizes, int n_in,
                              void* d_out, int out_size, void* d_ws, size_t ws_size,
                              hipStream_t stream) {
  const float* x     = (const float*)d_in[0];
  const float* gamma = (const float*)d_in[1];
  const float* beta  = (const float*)d_in[2];
  const float* wq    = (const float*)d_in[3];
  const float* bq    = (const float*)d_in[4];
  const float* wk    = (const float*)d_in[5];
  const float* bk    = (const float*)d_in[6];
  const float* wv    = (const float*)d_in[7];
  const float* bv    = (const float*)d_in[8];
  const float* wo    = (const float*)d_in[9];
  const float* bo    = (const float*)d_in[10];
  float* out = (float*)d_out;

  const size_t T = (size_t)Bn * Cn * Nsp;
  _Float16* xnP  = (_Float16*)d_ws;
  _Float16* qP   = xnP  + T;
  _Float16* kP   = qP   + T;
  _Float16* vP   = kP   + T;
  _Float16* attP = vP   + T;
  _Float16* wqP  = attP + T;
  _Float16* wkP  = wqP  + (size_t)Cn * Cn;
  _Float16* wvP  = wkP  + (size_t)Cn * Cn;
  _Float16* woP  = wvP  + (size_t)Cn * Cn;

  prep_weights_kernel<<<dim3(8, 16), 32, 0, stream>>>(
      wq, wk, wv, wo, wqP, wkP, wvP, woP);

  gn_kernel<<<Bn * GRP, 256, 0, stream>>>(x, gamma, beta, xnP);

  qkv_kernel<<<dim3(Nsp / 16, Cn / 16, Bn), 32, 0, stream>>>(
      xnP, wqP, wkP, wvP, bq, bk, bv, qP, kP, vP);

  attn_kernel<<<dim3(Nsp / 64, Bn), 256, 0, stream>>>(qP, kP, vP, attP);

  proj_kernel<<<dim3(Nsp / 16, Cn / 16, Bn), 32, 0, stream>>>(
      attP, woP, bo, x, out);
}
